// Codred_16801912062516
// MI455X (gfx1250) — matrix operand order repaired and checked by reading, not compiled, not run on hardware
//
#include <hip/hip_runtime.h>
#include <hip/hip_bf16.h>

#define D_    768
#define H_    8
#define DK_   96
#define L_    4
#define FF_   1024
#define NREL_ 97
#define BAG_  8
#define SEQ_  512
#define S_    5
#define E_    40
#define T_    1600   // E_*E_

typedef __attribute__((ext_vector_type(16))) __bf16 v16bf;
typedef __attribute__((ext_vector_type(8)))  float  v8f;
typedef unsigned short ushort_t;

union Frag {
  v16bf bf;
  unsigned int u[8];
};

__device__ __forceinline__ ushort_t f2bf(float f) {
  unsigned int u = __float_as_uint(f);
  u += 0x7FFFu + ((u >> 16) & 1u);   // round-to-nearest-even
  return (ushort_t)(u >> 16);
}

__device__ __forceinline__ v8f wmma_bf16(v16bf a, v16bf b, v8f c) {
  return __builtin_amdgcn_wmma_f32_16x16x32_bf16(
      false, a, false, b, (short)0, c, false, false);
}

// 16-byte async copy global -> LDS (CDNA5 GLOBAL_LOAD_ASYNC_TO_LDS_B128, ASYNCcnt)
__device__ __forceinline__ void async_cp16(void* lds_ptr, const void* gptr) {
  unsigned lds_off = (unsigned)(size_t)lds_ptr;   // generic LDS addr: low 32 bits = LDS offset
  asm volatile("global_load_async_to_lds_b128 %0, %1, off"
               :: "v"(lds_off), "v"(gptr) : "memory");
}
// Wait until all of this wave's async copies have landed.
__device__ __forceinline__ void async_wait0() {
  asm volatile("s_wait_asynccnt 0" ::: "memory");
}
// Double-buffer wait: allow the newest 6 copies (next tile) to remain in flight;
// async loads complete in order, so the previous tile's 6 copies are done.
__device__ __forceinline__ void async_wait6() {
  asm volatile("s_wait_asynccnt 6" ::: "memory");
}

// ---------------------------------------------------------------------------
// Weight pre-pass: WT[n*K + k] = bf16(W[k*N + n])  (transpose + convert, once)
// ---------------------------------------------------------------------------
__global__ void convT_k(const float* __restrict__ W, ushort_t* __restrict__ WT,
                        int K, int N) {
  const int idx = blockIdx.x * 256 + threadIdx.x;
  if (idx < K * N) {
    const int n = idx / K, k = idx % K;
    WT[idx] = f2bf(W[(size_t)k * N + n]);
  }
}

// ---------------------------------------------------------------------------
// Span max-pool -> bf16 htb [E,D]
// ---------------------------------------------------------------------------
__global__ void span_pool_k(const float* __restrict__ emb, const int* __restrict__ spans,
                            ushort_t* __restrict__ htb) {
  const int e = blockIdx.x;
  const int bag = e / S_;
  const int s0 = spans[e * 2 + 0];
  const int s1 = spans[e * 2 + 1];
#pragma unroll
  for (int c = 0; c < 3; ++c) {
    const int d = threadIdx.x + c * 256;
    float m = -1e30f;
    for (int p = s0; p <= s1; ++p)
      m = fmaxf(m, emb[((size_t)bag * SEQ_ + p) * D_ + d]);
    htb[(size_t)e * D_ + d] = f2bf(m);
  }
}

// ---------------------------------------------------------------------------
// alpha[i*E+j][d] = bf16(relu(u[i][d] + v[j][d]))
// ---------------------------------------------------------------------------
__global__ void alpha_k(const float* __restrict__ u, const float* __restrict__ v,
                        ushort_t* __restrict__ out) {
  const int t = blockIdx.x;
  const int i = t / E_, j = t % E_;
#pragma unroll
  for (int c = 0; c < 3; ++c) {
    const int d = threadIdx.x + c * 256;
    out[(size_t)t * D_ + d] = f2bf(fmaxf(u[(size_t)i * D_ + d] + v[(size_t)j * D_ + d], 0.f));
  }
}

// ---------------------------------------------------------------------------
// bf16 WMMA GEMM: C[M,N] = act(A[M,K] @ BT[N,K]^T + bias[N])
// A, BT bf16; LDS double-buffered via async copies pipelined on ASYNCcnt.
// Block tile 64(M) x 128(N), 256 threads = 8 waves (2x4), each wave 32x32.
// flags: 1=relu, 2=write f32 Cf, 4=write bf16 Cb, 8=Cb transposed [N,M]
// ---------------------------------------------------------------------------
__global__ void gemm_bb_k(const ushort_t* __restrict__ A, const ushort_t* __restrict__ BT,
                          const float* __restrict__ bias,
                          float* __restrict__ Cf, ushort_t* __restrict__ Cb,
                          int M, int N, int K, int flags) {
  __shared__ __align__(16) ushort_t sA[2][64 * 64];    // [m][k]
  __shared__ __align__(16) ushort_t sB[2][128 * 64];   // [n][k]
  const int tid  = threadIdx.x;
  const int lane = tid & 31;
  const int wave = tid >> 5;
  const int half = (lane >> 4) & 1;
  const int ln15 = lane & 15;
  const int wr = wave >> 2;                 // 0..1  -> 32 rows
  const int wc = wave & 3;                  // 0..3  -> 32 cols
  const int m0 = blockIdx.x * 64;
  const int n0 = blockIdx.y * 128;

  // 6 async instructions per wave per tile (2 for A, 4 for BT)
  auto issue_tile = [&](int kt, int buf) {
#pragma unroll
    for (int i = 0; i < 2; ++i) {
      const int c = tid + i * 256;
      const int r = c >> 3;
      const int q8 = (c & 7) << 3;
      int gr = m0 + r; if (gr >= M) gr = M - 1;   // clamp; garbage rows never stored
      async_cp16(sA[buf] + r * 64 + q8, A + (size_t)gr * K + kt + q8);
    }
#pragma unroll
    for (int i = 0; i < 4; ++i) {
      const int c = tid + i * 256;
      const int n = c >> 3;
      const int q8 = (c & 7) << 3;
      async_cp16(sB[buf] + n * 64 + q8, BT + (size_t)(n0 + n) * K + kt + q8);
    }
  };

  v8f acc[2][2];
#pragma unroll
  for (int mt = 0; mt < 2; ++mt)
#pragma unroll
    for (int nt = 0; nt < 2; ++nt) acc[mt][nt] = {};

  const int nk = K >> 6;
  issue_tile(0, 0);
  for (int it = 0; it < nk; ++it) {
    const int cur = it & 1;
    const bool more = (it + 1) < nk;
    if (more) issue_tile((it + 1) << 6, cur ^ 1);   // prefetch next tile
    if (more) async_wait6(); else async_wait0();    // current tile landed
    __syncthreads();
    const ushort_t* pA = sA[cur];
    const ushort_t* pB = sB[cur];
#pragma unroll
    for (int ks = 0; ks < 2; ++ks) {
      const int kk0 = ks * 32;
      Frag af[2], bf2[2];
#pragma unroll
      for (int mt = 0; mt < 2; ++mt) {
        const int row = wr * 32 + mt * 16 + ln15;
#pragma unroll
        for (int w2 = 0; w2 < 8; ++w2) {
          const int kb = (w2 < 4 ? 2 * w2 : 16 + 2 * (w2 - 4)) + 8 * half;
          af[mt].u[w2] = *(const unsigned int*)(pA + row * 64 + kk0 + kb);
        }
      }
#pragma unroll
      for (int nt = 0; nt < 2; ++nt) {
        const int n = wc * 32 + nt * 16 + ln15;
        const int base = n * 64 + kk0;
#pragma unroll
        for (int w2 = 0; w2 < 8; ++w2) {
          const int kb = (w2 < 4 ? 2 * w2 : 16 + 2 * (w2 - 4)) + 8 * half;
          bf2[nt].u[w2] = *(const unsigned int*)(pB + base + kb);
        }
      }
#pragma unroll
      for (int mt = 0; mt < 2; ++mt)
#pragma unroll
        for (int nt = 0; nt < 2; ++nt)
          acc[mt][nt] = wmma_bf16(af[mt].bf, bf2[nt].bf, acc[mt][nt]);
    }
    __syncthreads();   // all reads of buf `cur` done before it is refilled
  }

#pragma unroll
  for (int mt = 0; mt < 2; ++mt) {
#pragma unroll
    for (int nt = 0; nt < 2; ++nt) {
#pragma unroll
      for (int r = 0; r < 8; ++r) {
        const int grow = m0 + wr * 32 + mt * 16 + r + 8 * half;
        const int gcol = n0 + wc * 32 + nt * 16 + ln15;
        if (grow < M) {
          float val = acc[mt][nt][r] + bias[gcol];
          if (flags & 1) val = fmaxf(val, 0.f);
          if (flags & 2) Cf[(size_t)grow * N + gcol] = val;
          if (flags & 4) {
            const ushort_t bv = f2bf(val);
            if (flags & 8) Cb[(size_t)gcol * M + grow] = bv;
            else           Cb[(size_t)grow * N + gcol] = bv;
          }
        }
      }
    }
  }
}

// ---------------------------------------------------------------------------
// Fused flash attention. q,k: bf16 [T,768]; vT: bf16 [768,T]; out: bf16 [T,768].
// Block = 4 waves; wave owns 32 query rows; key blocks of 32, K/V tiles
// double-buffered with async global->LDS copies pipelined on ASYNCcnt.
// ---------------------------------------------------------------------------
__global__ void flash_attn_k(const ushort_t* __restrict__ q, const ushort_t* __restrict__ k,
                             const ushort_t* __restrict__ vT, ushort_t* __restrict__ o) {
  __shared__ __align__(16) ushort_t sK[2][32 * 96];    // [key][dk]
  __shared__ __align__(16) ushort_t sV[2][96 * 32];    // [dk][key]
  __shared__ __align__(16) ushort_t sP[4][32 * 32];    // per wave: [m][key]
  const int tid  = threadIdx.x;
  const int lane = tid & 31;
  const int wave = tid >> 5;
  const int half = (lane >> 4) & 1;
  const int ln15 = lane & 15;
  const int h = blockIdx.y;
  const int qbase = blockIdx.x * 128 + wave * 32;
  const int hbase = h * DK_;
  const float scale = 0.1020620726f;   // 1/sqrt(96)

  // 6 async instructions per wave per key block (3 for K, 3 for V)
  auto issue_kv = [&](int kt0, int buf) {
#pragma unroll
    for (int i = 0; i < 3; ++i) {
      const int c = tid + i * 128;
      const int key = c / 12, ch = c % 12;
      async_cp16(sK[buf] + key * 96 + ch * 8, k + (size_t)(kt0 + key) * D_ + hbase + ch * 8);
    }
#pragma unroll
    for (int i = 0; i < 3; ++i) {
      const int c = tid + i * 128;
      const int dk = c >> 2, ch = c & 3;
      async_cp16(sV[buf] + dk * 32 + ch * 8, vT + (size_t)(hbase + dk) * T_ + kt0 + ch * 8);
    }
  };

  // Q fragments straight from global bf16 (pairs are contiguous u32 reads)
  Frag qf[2][3];
#pragma unroll
  for (int mt = 0; mt < 2; ++mt) {
    int row = qbase + mt * 16 + ln15;
    if (row >= T_) row = T_ - 1;
    const ushort_t* qrow = q + (size_t)row * D_ + hbase;
#pragma unroll
    for (int ks = 0; ks < 3; ++ks)
#pragma unroll
      for (int w2 = 0; w2 < 8; ++w2) {
        const int kb = ks * 32 + (w2 < 4 ? 2 * w2 : 16 + 2 * (w2 - 4)) + 8 * half;
        qf[mt][ks].u[w2] = *(const unsigned int*)(qrow + kb);
      }
  }

  v8f O[2][6];
#pragma unroll
  for (int mt = 0; mt < 2; ++mt)
#pragma unroll
    for (int nt = 0; nt < 6; ++nt) O[mt][nt] = {};
  float rowmax[2][8], rowsum[2][8];
#pragma unroll
  for (int mt = 0; mt < 2; ++mt)
#pragma unroll
    for (int r = 0; r < 8; ++r) { rowmax[mt][r] = -1e30f; rowsum[mt][r] = 0.f; }

  ushort_t* sPw = sP[wave];
  const int nblk = T_ / 32;

  issue_kv(0, 0);
  for (int blk = 0; blk < nblk; ++blk) {
    const int cur = blk & 1;
    const bool more = (blk + 1) < nblk;
    if (more) issue_kv((blk + 1) * 32, cur ^ 1);    // prefetch next key block
    if (more) async_wait6(); else async_wait0();
    __syncthreads();
    const ushort_t* pK = sK[cur];
    const ushort_t* pV = sV[cur];

    v8f S[2][2];
    S[0][0] = {}; S[0][1] = {}; S[1][0] = {}; S[1][1] = {};
#pragma unroll
    for (int ks = 0; ks < 3; ++ks) {
      Frag bfr[2];
#pragma unroll
      for (int nt = 0; nt < 2; ++nt) {
        const int key = nt * 16 + ln15;
        const int base = key * 96 + ks * 32;
#pragma unroll
        for (int w2 = 0; w2 < 8; ++w2) {
          const int kb = (w2 < 4 ? 2 * w2 : 16 + 2 * (w2 - 4)) + 8 * half;
          bfr[nt].u[w2] = *(const unsigned int*)(pK + base + kb);
        }
      }
#pragma unroll
      for (int mt = 0; mt < 2; ++mt)
#pragma unroll
        for (int nt = 0; nt < 2; ++nt)
          S[mt][nt] = wmma_bf16(qf[mt][ks].bf, bfr[nt].bf, S[mt][nt]);
    }

    // online softmax (reductions stay inside each 16-lane half; wave32)
#pragma unroll
    for (int mt = 0; mt < 2; ++mt) {
#pragma unroll
      for (int r = 0; r < 8; ++r) {
        const float v0 = S[mt][0][r] * scale;
        const float v1 = S[mt][1][r] * scale;
        float mx = fmaxf(v0, v1);
#pragma unroll
        for (int msk = 1; msk < 16; msk <<= 1)
          mx = fmaxf(mx, __shfl_xor(mx, msk, 32));
        const float nm = fmaxf(rowmax[mt][r], mx);
        const float corr = __expf(rowmax[mt][r] - nm);
        rowmax[mt][r] = nm;
        const float p0 = __expf(v0 - nm);
        const float p1 = __expf(v1 - nm);
        float ps = p0 + p1;
#pragma unroll
        for (int msk = 1; msk < 16; msk <<= 1)
          ps += __shfl_xor(ps, msk, 32);
        rowsum[mt][r] = rowsum[mt][r] * corr + ps;
#pragma unroll
        for (int nt = 0; nt < 6; ++nt) O[mt][nt][r] *= corr;
        const int m = mt * 16 + r + 8 * half;
        sPw[m * 32 + ln15]      = f2bf(p0);
        sPw[m * 32 + 16 + ln15] = f2bf(p1);
      }
    }
    __syncthreads();

    // O += P @ V   (one WMMA k-step of 32 keys)
    Frag vfr[6];
#pragma unroll
    for (int nt = 0; nt < 6; ++nt) {
      const int n = nt * 16 + ln15;
      const int base = n * 32;
#pragma unroll
      for (int w2 = 0; w2 < 8; ++w2) {
        const int kb = (w2 < 4 ? 2 * w2 : 16 + 2 * (w2 - 4)) + 8 * half;
        vfr[nt].u[w2] = *(const unsigned int*)(pV + base + kb);
      }
    }
#pragma unroll
    for (int mt = 0; mt < 2; ++mt) {
      Frag pf;
      const int m = mt * 16 + ln15;
#pragma unroll
      for (int w2 = 0; w2 < 8; ++w2) {
        const int kb = (w2 < 4 ? 2 * w2 : 16 + 2 * (w2 - 4)) + 8 * half;
        pf.u[w2] = *(const unsigned int*)(sPw + m * 32 + kb);
      }
#pragma unroll
      for (int nt = 0; nt < 6; ++nt)
        O[mt][nt] = wmma_bf16(pf.bf, vfr[nt].bf, O[mt][nt]);
    }
    __syncthreads();   // all reads of buf `cur` done before it is refilled
  }

#pragma unroll
  for (int mt = 0; mt < 2; ++mt) {
#pragma unroll
    for (int r = 0; r < 8; ++r) {
      const int row = qbase + mt * 16 + r + 8 * half;
      if (row < T_) {
        const float inv = 1.f / rowsum[mt][r];
#pragma unroll
        for (int nt = 0; nt < 6; ++nt)
          o[(size_t)row * D_ + hbase + nt * 16 + ln15] = f2bf(O[mt][nt][r] * inv);
      }
    }
  }
}

// ---------------------------------------------------------------------------
// x = LayerNorm(x + y)*g + b  (f32), plus bf16 shadow copy xb for next GEMMs
// ---------------------------------------------------------------------------
__global__ void add_ln_k(float* __restrict__ x, const float* __restrict__ y,
                         const float* __restrict__ g, const float* __restrict__ b,
                         ushort_t* __restrict__ xb) {
  __shared__ float red[256];
  const int t = blockIdx.x;
  const int tid = threadIdx.x;
  float val[3];
  float s = 0.f;
#pragma unroll
  for (int i = 0; i < 3; ++i) {
    const int d = tid + i * 256;
    val[i] = x[(size_t)t * D_ + d] + y[(size_t)t * D_ + d];
    s += val[i];
  }
  red[tid] = s; __syncthreads();
  for (int off = 128; off > 0; off >>= 1) {
    if (tid < off) red[tid] += red[tid + off];
    __syncthreads();
  }
  const float mean = red[0] * (1.f / (float)D_);
  __syncthreads();
  float s2 = 0.f;
#pragma unroll
  for (int i = 0; i < 3; ++i) { const float dd = val[i] - mean; s2 += dd * dd; }
  red[tid] = s2; __syncthreads();
  for (int off = 128; off > 0; off >>= 1) {
    if (tid < off) red[tid] += red[tid + off];
    __syncthreads();
  }
  const float inv = rsqrtf(red[0] * (1.f / (float)D_) + 1e-5f);
#pragma unroll
  for (int i = 0; i < 3; ++i) {
    const int d = tid + i * 256;
    const float r = (val[i] - mean) * inv * g[d] + b[d];
    x[(size_t)t * D_ + d]  = r;
    xb[(size_t)t * D_ + d] = f2bf(r);
  }
}

// ---------------------------------------------------------------------------
// Final gather + predict + bag max -> out[97]
// ---------------------------------------------------------------------------
__global__ void predict_k(const float* __restrict__ x, const float* __restrict__ w,
                          const float* __restrict__ b, float* __restrict__ out) {
  const int r = threadIdx.x;
  if (r >= NREL_) return;
  float best = -1e30f;
  for (int bag = 0; bag < BAG_; ++bag) {
    const int tok = 205 * bag + 1;   // (5*bag)*40 + (5*bag+1)
    float acc = b[r];
    for (int kk = 0; kk < D_; ++kk)
      acc += x[(size_t)tok * D_ + kk] * w[(size_t)kk * NREL_ + r];
    best = fmaxf(best, acc);
  }
  out[r] = best;
}

// ---------------------------------------------------------------------------
static inline void gemm(const ushort_t* A, const ushort_t* BT, const float* bias,
                        float* Cf, ushort_t* Cb, int M, int N, int K, int flags,
                        hipStream_t s) {
  dim3 grid((M + 63) / 64, N / 128);
  gemm_bb_k<<<grid, 256, 0, s>>>(A, BT, bias, Cf, Cb, M, N, K, flags);
}

static inline void convT(const float* W, ushort_t* WT, int K, int N, hipStream_t s) {
  convT_k<<<(K * N + 255) / 256, 256, 0, s>>>(W, WT, K, N);
}

extern "C" void kernel_launch(void* const* d_in, const int* in_sizes, int n_in,
                              void* d_out, int out_size, void* d_ws, size_t ws_size,
                              hipStream_t stream) {
  (void)in_sizes; (void)n_in; (void)out_size; (void)ws_size;
  const float* emb   = (const float*)d_in[0];
  const int*   spans = (const int*)d_in[1];
  const float* wu_w  = (const float*)d_in[2];
  const float* wu_b  = (const float*)d_in[3];
  const float* wvv_w = (const float*)d_in[4];
  const float* wvv_b = (const float*)d_in[5];
  const float* lnr_w = (const float*)d_in[6];
  const float* lnr_b = (const float*)d_in[7];
  const float* wq = (const float*)d_in[8];
  const float* bq = (const float*)d_in[9];
  const float* wk = (const float*)d_in[10];
  const float* bk = (const float*)d_in[11];
  const float* wv = (const float*)d_in[12];
  const float* bv = (const float*)d_in[13];
  const float* wo = (const float*)d_in[14];
  const float* bo = (const float*)d_in[15];
  const float* w1 = (const float*)d_in[16];
  const float* b1 = (const float*)d_in[17];
  const float* w2 = (const float*)d_in[18];
  const float* b2 = (const float*)d_in[19];
  const float* g1  = (const float*)d_in[20];
  const float* be1 = (const float*)d_in[21];
  const float* g2  = (const float*)d_in[22];
  const float* be2 = (const float*)d_in[23];
  const float* pw = (const float*)d_in[24];
  const float* pb = (const float*)d_in[25];
  float* out = (float*)d_out;

  // ---- workspace layout (byte offsets) ----
  char* base = (char*)d_ws;
  float*    u_      = (float*)(base + 0);           // 40*768 f32
  float*    v_      = (float*)(base + 122880);      // 40*768 f32
  float*    x_      = (float*)(base + 245760);      // 1600*768 f32
  float*    tmp_    = (float*)(base + 5160960);     // 1600*768 f32
  ushort_t* htb_b   = (ushort_t*)(base + 10076160); // 40*768 bf16
  ushort_t* alpha_b = (ushort_t*)(base + 10137600); // 1600*768 bf16
  ushort_t* xb      = (ushort_t*)(base + 12595200); // 1600*768 bf16
  ushort_t* qb      = (ushort_t*)(base + 15052800); // 1600*768 bf16
  ushort_t* kb      = (ushort_t*)(base + 17510400); // 1600*768 bf16
  ushort_t* vbT     = (ushort_t*)(base + 19968000); // 768*1600 bf16 (transposed)
  ushort_t* aob     = (ushort_t*)(base + 22425600); // 1600*768 bf16
  ushort_t* ff1b    = (ushort_t*)(base + 24883200); // 1600*1024 bf16
  ushort_t* wuT     = (ushort_t*)(base + 28160000); // 768*768 bf16 each
  ushort_t* wvvT    = (ushort_t*)(base + 29339648);
  ushort_t* lnrT    = (ushort_t*)(base + 30519296);
  ushort_t* wLT     = (ushort_t*)(base + 31698944); // per-layer packed weights
  const size_t DD = (size_t)D_ * D_;                // 589824
  const size_t DF = (size_t)D_ * FF_;               // 786432
  const size_t LSTR = 4 * DD + 2 * DF;              // bf16 elems per layer

  // ---- weight pre-pass: transpose + convert to bf16 (once per launch) ----
  convT(wu_w,  wuT,  D_, D_, stream);
  convT(wvv_w, wvvT, D_, D_, stream);
  convT(lnr_w, lnrT, D_, D_, stream);
  for (int l = 0; l < L_; ++l) {
    ushort_t* p = wLT + (size_t)l * LSTR;
    convT(wq + (size_t)l * DD, p + 0 * DD, D_, D_, stream);
    convT(wk + (size_t)l * DD, p + 1 * DD, D_, D_, stream);
    convT(wv + (size_t)l * DD, p + 2 * DD, D_, D_, stream);
    convT(wo + (size_t)l * DD, p + 3 * DD, D_, D_, stream);
    convT(w1 + (size_t)l * DF, p + 4 * DD,      D_,  FF_, stream);  // [FF,768]
    convT(w2 + (size_t)l * DF, p + 4 * DD + DF, FF_, D_,  stream);  // [768,1024]
  }

  // ---- pipeline ----
  span_pool_k<<<E_, 256, 0, stream>>>(emb, spans, htb_b);
  gemm(htb_b, wuT,  wu_b,  u_, nullptr, E_, D_, D_, 2, stream);
  gemm(htb_b, wvvT, wvv_b, v_, nullptr, E_, D_, D_, 2, stream);
  alpha_k<<<T_, 256, 0, stream>>>(u_, v_, alpha_b);
  gemm(alpha_b, lnrT, lnr_b, x_, xb, T_, D_, D_, 1 | 2 | 4, stream);

  for (int l = 0; l < L_; ++l) {
    ushort_t* p = wLT + (size_t)l * LSTR;
    gemm(xb, p + 0 * DD, bq + l * D_, nullptr, qb,  T_, D_, D_, 4,     stream);
    gemm(xb, p + 1 * DD, bk + l * D_, nullptr, kb,  T_, D_, D_, 4,     stream);
    gemm(xb, p + 2 * DD, bv + l * D_, nullptr, vbT, T_, D_, D_, 4 | 8, stream);
    flash_attn_k<<<dim3((T_ + 127) / 128, H_), 128, 0, stream>>>(qb, kb, vbT, aob);
    gemm(aob, p + 3 * DD, bo + l * D_, tmp_, nullptr, T_, D_, D_, 2, stream);
    add_ln_k<<<T_, 256, 0, stream>>>(x_, tmp_, g1 + l * D_, be1 + l * D_, xb);
    gemm(xb, p + 4 * DD, b1 + l * FF_, nullptr, ff1b, T_, FF_, D_, 1 | 4, stream);
    gemm(ff1b, p + 4 * DD + DF, b2 + l * D_, tmp_, nullptr, T_, D_, FF_, 2, stream);
    add_ln_k<<<T_, 256, 0, stream>>>(x_, tmp_, g2 + l * D_, be2 + l * D_, xb);
  }
  predict_k<<<1, 128, 0, stream>>>(x_, pw, pb, out);
}
